// DDINetworkEncoder_78855599555023
// MI455X (gfx1250) — compile-verified, matrix operand order checked by reading, not run on hardware
//
#include <hip/hip_runtime.h>
#include <hip/hip_bf16.h>

typedef __attribute__((ext_vector_type(16))) _Float16 v16h;
typedef __attribute__((ext_vector_type(8)))  _Float16 v8h;
typedef __attribute__((ext_vector_type(8)))  float    v8f;

union F16x16 { v16h v; v8h g[2]; };

// ---------------------------------------------------------------------------
// Degree / normalization (deg -> dinv in place)
// ---------------------------------------------------------------------------
__global__ void deg_init_kernel(float* __restrict__ deg, int n) {
    int i = blockIdx.x * blockDim.x + threadIdx.x;
    if (i < n) deg[i] = 1.0f;                       // self-loop weight
}

__global__ void deg_edges_kernel(const int* __restrict__ ei,
                                 const float* __restrict__ ew,
                                 float* __restrict__ deg, int E) {
    int e = blockIdx.x * blockDim.x + threadIdx.x;
    if (e < E) unsafeAtomicAdd(&deg[ei[E + e]], ew[e]);   // dst side
}

__global__ void deg_rsqrt_kernel(float* __restrict__ deg, int n) {
    int i = blockIdx.x * blockDim.x + threadIdx.x;
    if (i < n) {
        float d = deg[i];
        deg[i] = (d > 0.0f) ? rsqrtf(d) : 0.0f;
    }
}

// ---------------------------------------------------------------------------
// Pack x [N,10] f32 -> xp [N,16] f16, zero-padded (clamped address, no branch)
// ---------------------------------------------------------------------------
__global__ void pack_x_kernel(const float* __restrict__ x,
                              _Float16* __restrict__ xp, int n16) {
    int i = blockIdx.x * blockDim.x + threadIdx.x;
    if (i < n16) {
        int row = i >> 4, k = i & 15;
        int kc = (k < 10) ? k : 9;
        float v = x[row * 10 + kc];
        xp[i] = (k < 10) ? (_Float16)v : (_Float16)0.0f;
    }
}

// ---------------------------------------------------------------------------
// WMMA GEMM: out = rowgather(A) @ W (+ bias), optional fused self-loop term
//   A: f16 [*, KP] (KP = 16 or 64, zero-padded), rows via rowidx if GATHER
//   W: f32 [Kreal, NOUT] -> staged transposed f16 [NOUT, KP] in LDS
//   out: f32 or f16, [nrows, NOUT]; nrows must be a multiple of 16
//   SELFINIT: also write agg[row,:] = c[row,:] * dinv[row]^2 (GCN self loop)
// One wave computes a 16 x NOUT slab with v_wmma_f32_16x16x32_f16.
// Fragment loads are 16B runs: global_load_b128 (A) / ds_load_b128 (B).
// ---------------------------------------------------------------------------
template <int KP, int NOUT, bool GATHER, bool OUT_F16, bool BIAS, bool SELFINIT>
__global__ void wmma_gemm_kernel(const _Float16* __restrict__ A,
                                 const int* __restrict__ rowidx,
                                 const float* __restrict__ W,
                                 const float* __restrict__ bias,
                                 void* __restrict__ outv,
                                 float* __restrict__ agg,
                                 const float* __restrict__ dinv,
                                 int nrows, int Kreal) {
    __shared__ __align__(16) _Float16 Wt[NOUT * KP];   // W^T, f16, zero-padded

    // Cooperative stage: Wt[n][k] = (k < Kreal) ? W[k][n] : 0
    for (int idx = threadIdx.x; idx < NOUT * KP; idx += blockDim.x) {
        int n = idx / KP;
        int k = idx - n * KP;
        int ks = (k < Kreal) ? k : (Kreal - 1);      // clamp -> unconditional load
        float w = W[ks * NOUT + n];
        Wt[idx] = (k < Kreal) ? (_Float16)w : (_Float16)0.0f;
    }
    __syncthreads();

    const int lane = threadIdx.x & 31;
    const int wave = threadIdx.x >> 5;
    const int tile = blockIdx.x * (blockDim.x >> 5) + wave;
    const int r0   = tile * 16;
    if (r0 >= nrows) return;                          // wave-uniform

    const int half = lane >> 4;                       // lane-group (K interleave)
    const int l15  = lane & 15;

    const long arow = GATHER ? (long)rowidx[r0 + l15] : (long)(r0 + l15);
    const _Float16* __restrict__ Arow = A + arow * KP;

    constexpr int  KCH      = (KP + 31) / 32;         // 1 (KP=16) or 2 (KP=64)
    constexpr bool HI_VALID = (KP >= 32);             // group2 inside padded row?

    // A fragments: ISA 16-bit A layout = two contiguous 8xf16 runs per chunk
    v16h areg[KCH];
#pragma unroll
    for (int kc = 0; kc < KCH; ++kc) {
        F16x16 a;
        a.g[0] = *(const v8h*)(Arow + kc * 32 + 8 * half);
        if (HI_VALID) a.g[1] = *(const v8h*)(Arow + kc * 32 + 16 + 8 * half);
        else          a.g[1] = (v8h){};
        areg[kc] = a.v;
    }

    // Self-loop scale per output row (hoisted out of the n-tile loop)
    float dsq[8];
    if (SELFINIT) {
#pragma unroll
        for (int r = 0; r < 8; ++r) {
            float d = dinv[r0 + r + half * 8];
            dsq[r] = d * d;
        }
    }

    float* outF = (float*)outv;
    _Float16* outH = (_Float16*)outv;

#pragma unroll
    for (int nt = 0; nt < NOUT / 16; ++nt) {
        const _Float16* Wn = Wt + (nt * 16 + l15) * KP;  // lane -> column of W
        v8f c = {};
#pragma unroll
        for (int kc = 0; kc < KCH; ++kc) {
            F16x16 b;
            b.g[0] = *(const v8h*)(Wn + kc * 32 + 8 * half);
            if (HI_VALID) b.g[1] = *(const v8h*)(Wn + kc * 32 + 16 + 8 * half);
            else          b.g[1] = (v8h){};
            c = __builtin_amdgcn_wmma_f32_16x16x32_f16(
                    false, areg[kc], false, b.v, (short)0, c, false, false);
        }
        float bn = 0.0f;
        if (BIAS) bn = bias[nt * 16 + l15];
        // C/D layout: VGPR r -> row = r0 + r + 8*half, col = nt*16 + l15
#pragma unroll
        for (int r = 0; r < 8; ++r) {
            long off = (long)(r0 + r + half * 8) * NOUT + nt * 16 + l15;
            float val = c[r] + bn;
            if (OUT_F16) outH[off] = (_Float16)val;
            else         outF[off] = val;
            if (SELFINIT) agg[off] = val * dsq[r];
        }
    }
}

// ---------------------------------------------------------------------------
// GCN aggregation
// ---------------------------------------------------------------------------
// 16 threads per edge; each moves one float4 of the 64-wide feature row
__global__ void edge_scatter_kernel(const int* __restrict__ ei,
                                    const float* __restrict__ ew,
                                    const float* __restrict__ dinv,
                                    const float* __restrict__ t,
                                    float* __restrict__ agg, int E) {
    int tid   = blockIdx.x * blockDim.x + threadIdx.x;
    int e     = tid >> 4;
    int chunk = tid & 15;
    if (e >= E) return;
    int src = ei[e];
    int dst = ei[E + e];
    float norm = dinv[src] * ew[e] * dinv[dst];
    const float4 v = *(const float4*)(t + (long)src * 64 + chunk * 4);
    float* o = agg + (long)dst * 64 + chunk * 4;
    unsafeAtomicAdd(o + 0, v.x * norm);
    unsafeAtomicAdd(o + 1, v.y * norm);
    unsafeAtomicAdd(o + 2, v.z * norm);
    unsafeAtomicAdd(o + 3, v.w * norm);
}

// h16[i] = f16( relu(agg[i] + bias[col]) )
__global__ void bias_relu_f16_kernel(const float* __restrict__ agg,
                                     const float* __restrict__ bias,
                                     _Float16* __restrict__ h16, int n64) {
    int i = blockIdx.x * blockDim.x + threadIdx.x;
    if (i < n64) h16[i] = (_Float16)fmaxf(agg[i] + bias[i & 63], 0.0f);
}

// ---------------------------------------------------------------------------
// Launcher
// ---------------------------------------------------------------------------
extern "C" void kernel_launch(void* const* d_in, const int* in_sizes, int n_in,
                              void* d_out, int out_size, void* d_ws, size_t ws_size,
                              hipStream_t stream) {
    const float* x       = (const float*)d_in[0];    // [N,10]
    const int*   ei      = (const int*)  d_in[1];    // [2,E]
    const float* ew      = (const float*)d_in[2];    // [E]
    const int*   drug    = (const int*)  d_in[3];    // [B]
    const float* W_enc   = (const float*)d_in[4];    // [10,64]
    const float* b_enc   = (const float*)d_in[5];
    const float* W_conv1 = (const float*)d_in[6];    // [64,64]
    const float* b_conv1 = (const float*)d_in[7];
    const float* W_conv2 = (const float*)d_in[8];
    const float* b_conv2 = (const float*)d_in[9];
    const float* W_out   = (const float*)d_in[10];   // [64,128]
    const float* b_out   = (const float*)d_in[11];
    float* out = (float*)d_out;                      // [B,128]

    const int N   = in_sizes[0] / 10;
    const int E   = in_sizes[2];
    const int B   = in_sizes[3];
    const int n64 = N * 64;

    // Workspace: [dinv N f32][bufT N*64 f32][bufAgg N*64 f32][hA16 N*64 f16][xp16 N*16 f16]
    float*     dinv   = (float*)d_ws;
    float*     bufT   = dinv + N;
    float*     bufAgg = bufT + (long)n64;
    _Float16*  hA16   = (_Float16*)(bufAgg + (long)n64);
    _Float16*  xp16   = hA16 + (long)n64;

    const int TB = 256;
    const int gemm_grid_N = ((N / 16) + 7) / 8;      // 8 waves per block
    const int gemm_grid_B = ((B / 16) + 7) / 8;

    // --- degree -> dinv ---
    deg_init_kernel<<<(N + TB - 1) / TB, TB, 0, stream>>>(dinv, N);
    deg_edges_kernel<<<(E + TB - 1) / TB, TB, 0, stream>>>(ei, ew, dinv, E);
    deg_rsqrt_kernel<<<(N + TB - 1) / TB, TB, 0, stream>>>(dinv, N);

    // --- encoder: hA16 = f16(x @ W_enc + b_enc) ---
    pack_x_kernel<<<(N * 16 + TB - 1) / TB, TB, 0, stream>>>(x, xp16, N * 16);
    wmma_gemm_kernel<16, 64, false, true, true, false>
        <<<gemm_grid_N, TB, 0, stream>>>(xp16, nullptr, W_enc, b_enc,
                                         hA16, nullptr, nullptr, N, 10);

    // --- conv1: GEMM (+fused self-loop init) -> edge scatter -> bias+relu ---
    wmma_gemm_kernel<64, 64, false, false, false, true>
        <<<gemm_grid_N, TB, 0, stream>>>(hA16, nullptr, W_conv1, nullptr,
                                         bufT, bufAgg, dinv, N, 64);
    edge_scatter_kernel<<<(int)(((long)E * 16 + TB - 1) / TB), TB, 0, stream>>>(
        ei, ew, dinv, bufT, bufAgg, E);
    bias_relu_f16_kernel<<<(n64 + TB - 1) / TB, TB, 0, stream>>>(bufAgg, b_conv1, hA16, n64);

    // --- conv2 ---
    wmma_gemm_kernel<64, 64, false, false, false, true>
        <<<gemm_grid_N, TB, 0, stream>>>(hA16, nullptr, W_conv2, nullptr,
                                         bufT, bufAgg, dinv, N, 64);
    edge_scatter_kernel<<<(int)(((long)E * 16 + TB - 1) / TB), TB, 0, stream>>>(
        ei, ew, dinv, bufT, bufAgg, E);
    bias_relu_f16_kernel<<<(n64 + TB - 1) / TB, TB, 0, stream>>>(bufAgg, b_conv2, hA16, n64);

    // --- output projection on gathered rows only: out = hA16[drug] @ W_out + b_out ---
    wmma_gemm_kernel<64, 128, true, false, true, false>
        <<<gemm_grid_B, TB, 0, stream>>>(hA16, drug, W_out, b_out,
                                         out, nullptr, nullptr, B, 64);
}
